// PadMoE_23983097381217
// MI455X (gfx1250) — matrix operand order repaired and checked by reading, not compiled, not run on hardware
//
#include <hip/hip_runtime.h>
#include <hip/hip_bf16.h>

// Problem constants (from reference)
#define N_TOK 2048      // B*S
#define D_IN  768
#define E_EXP 8
#define O_OUT 3072

typedef __attribute__((ext_vector_type(16))) _Float16 v16h;
typedef __attribute__((ext_vector_type(8)))  _Float16 v8h;
typedef __attribute__((ext_vector_type(8)))  float    v8f;
typedef __attribute__((ext_vector_type(4)))  float    v4f;

// ---- workspace layout (bytes), all 256B-aligned ----
#define WS_CNT    0                                   // 8 x int
#define WS_IMP    256                                 // 8 x float
#define WS_LOAD   512                                 // 8 x float
#define WS_ELIST  1024                                // E*N x int   (65536 B)
#define WS_EG     (WS_ELIST + E_EXP * N_TOK * 4)      // E*N x float (65536 B)
#define WS_XH     (WS_EG    + E_EXP * N_TOK * 4)      // N*D   f16   (3 MB)
#define WS_WH     (WS_XH + N_TOK * D_IN * 2)          // E*O*D f16   (37.7 MB)
// total ~41 MB

// ------------------------------------------------------------------
// Kernel 0: zero y output region and the per-expert counters
// ------------------------------------------------------------------
__global__ __launch_bounds__(256)
void moe_zero_kernel(float* __restrict__ y, int* __restrict__ cnt,
                     float* __restrict__ imp, float* __restrict__ loadv) {
    size_t i = (size_t)blockIdx.x * blockDim.x + threadIdx.x;   // i < 1572864
    v4f z = {0.f, 0.f, 0.f, 0.f};
    ((v4f*)y)[i] = z;                                           // 6291456 floats total
    if (blockIdx.x == 0 && threadIdx.x < E_EXP) {
        cnt[threadIdx.x]   = 0;
        imp[threadIdx.x]   = 0.f;
        loadv[threadIdx.x] = 0.f;
    }
}

// ------------------------------------------------------------------
// Kernel 1a: Xh = f16(x)
// ------------------------------------------------------------------
__global__ __launch_bounds__(256)
void moe_cvt_x_kernel(const float* __restrict__ x, _Float16* __restrict__ xh) {
    size_t i = ((size_t)blockIdx.x * blockDim.x + threadIdx.x) * 8;
    v4f a = *(const v4f*)(x + i);
    v4f b = *(const v4f*)(x + i + 4);
    v8h h;
    h[0] = (_Float16)a[0]; h[1] = (_Float16)a[1];
    h[2] = (_Float16)a[2]; h[3] = (_Float16)a[3];
    h[4] = (_Float16)b[0]; h[5] = (_Float16)b[1];
    h[6] = (_Float16)b[2]; h[7] = (_Float16)b[3];
    *(v8h*)(xh + i) = h;
}

// ------------------------------------------------------------------
// Kernel 1b: Wh[e][o][d] = f16(expert_w[e][o][d] + static_w[o][d])
// ------------------------------------------------------------------
__global__ __launch_bounds__(256)
void moe_cvt_w_kernel(const float* __restrict__ ew, const float* __restrict__ sw,
                      _Float16* __restrict__ wh) {
    size_t i  = ((size_t)blockIdx.x * blockDim.x + threadIdx.x) * 8;
    size_t od = i % ((size_t)O_OUT * D_IN);     // matching static_w offset
    v4f a  = *(const v4f*)(ew + i);
    v4f a2 = *(const v4f*)(ew + i + 4);
    v4f s  = *(const v4f*)(sw + od);
    v4f s2 = *(const v4f*)(sw + od + 4);
    v8h h;
    h[0] = (_Float16)(a[0]  + s[0]);  h[1] = (_Float16)(a[1]  + s[1]);
    h[2] = (_Float16)(a[2]  + s[2]);  h[3] = (_Float16)(a[3]  + s[3]);
    h[4] = (_Float16)(a2[0] + s2[0]); h[5] = (_Float16)(a2[1] + s2[1]);
    h[6] = (_Float16)(a2[2] + s2[2]); h[7] = (_Float16)(a2[3] + s2[3]);
    *(v8h*)(wh + i) = h;
}

// ------------------------------------------------------------------
// Kernel 2: gating — logits = x @ w_gate, top-2 softmax, build
// per-expert token lists and importance/load accumulators.
// One 256-thread block per token.
// ------------------------------------------------------------------
__global__ __launch_bounds__(256)
void moe_gate_kernel(const float* __restrict__ x, const float* __restrict__ wg,
                     int* __restrict__ cnt, float* __restrict__ imp,
                     float* __restrict__ loadv, int* __restrict__ elist,
                     float* __restrict__ eg) {
    __shared__ float sm[256];
    const int n = blockIdx.x;
    const int t = threadIdx.x;
    const int e = t & 7;        // expert this thread contributes to
    const int c = t >> 3;       // chunk 0..31, 24 elements each
    const float* xr = x + (size_t)n * D_IN + c * 24;
    const float* wr = wg + (size_t)c * 24 * E_EXP + e;
    float p = 0.f;
    #pragma unroll
    for (int j = 0; j < 24; ++j)
        p += xr[j] * wr[(size_t)j * E_EXP];
    sm[t] = p;
    __syncthreads();
    for (int s = 128; s >= 8; s >>= 1) {
        if (t < s) sm[t] += sm[t + s];
        __syncthreads();
    }
    if (t == 0) {
        // top-2 (ties -> lowest index, matching lax.top_k)
        float v0 = -1e30f; int e0 = 0;
        #pragma unroll
        for (int k = 0; k < E_EXP; ++k)
            if (sm[k] > v0) { v0 = sm[k]; e0 = k; }
        float v1 = -1e30f; int e1 = 0;
        #pragma unroll
        for (int k = 0; k < E_EXP; ++k)
            if (k != e0 && sm[k] > v1) { v1 = sm[k]; e1 = k; }
        // softmax over {v0, v1}, v0 >= v1
        float ex = __expf(v1 - v0);
        float inv = 1.f / (1.f + ex);
        float g0 = inv;
        float g1 = ex * inv;
        int p0 = atomicAdd(&cnt[e0], 1);
        elist[e0 * N_TOK + p0] = n;  eg[e0 * N_TOK + p0] = g0;
        int p1 = atomicAdd(&cnt[e1], 1);
        elist[e1 * N_TOK + p1] = n;  eg[e1 * N_TOK + p1] = g1;
        atomicAdd(&imp[e0], g0);     atomicAdd(&imp[e1], g1);
        atomicAdd(&loadv[e0], 1.f);  atomicAdd(&loadv[e1], 1.f);
    }
}

// ------------------------------------------------------------------
// Kernel 3: loss = 0.01 * (cv2(importance) + cv2(load)), ddof=1
// ------------------------------------------------------------------
__global__ void moe_loss_kernel(const float* __restrict__ imp,
                                const float* __restrict__ loadv,
                                float* __restrict__ loss_out) {
    float mi = 0.f, ml = 0.f;
    for (int k = 0; k < E_EXP; ++k) { mi += imp[k]; ml += loadv[k]; }
    mi *= (1.f / E_EXP); ml *= (1.f / E_EXP);
    float vi = 0.f, vl = 0.f;
    for (int k = 0; k < E_EXP; ++k) {
        float di = imp[k] - mi;   vi += di * di;
        float dl = loadv[k] - ml; vl += dl * dl;
    }
    vi *= (1.f / (E_EXP - 1)); vl *= (1.f / (E_EXP - 1));
    *loss_out = 0.01f * (vi / (mi * mi + 1e-10f) + vl / (ml * ml + 1e-10f));
}

// ------------------------------------------------------------------
// Kernel 4: sparse expert GEMM via WMMA f16 -> f32.
// One wave computes a 32-token x 64-output tile for one expert:
// 2 A fragments reused over 4 B fragments = 8 WMMAs per k-step
// (21.3 flop/byte from L2 vs 12.8 for a 16-row tile).
// grid = (O/64, N/32, E); early-exit on token-tile past cnt[e].
// ------------------------------------------------------------------
__global__ __launch_bounds__(32)
void moe_gemm_kernel(const _Float16* __restrict__ xh, const _Float16* __restrict__ wh,
                     const float* __restrict__ eb, const int* __restrict__ cnt,
                     const int* __restrict__ elist, const float* __restrict__ eg,
                     float* __restrict__ y) {
    const int e  = blockIdx.z;
    const int mt = blockIdx.y;
    const int ob = blockIdx.x * 64;
    const int ce = cnt[e];
    if (mt * 32 >= ce) return;               // wave-uniform exit

    __shared__ int   tokLds[32];
    __shared__ float gLds[32];
    const int lane = threadIdx.x;
    {
        int idx = mt * 32 + lane;
        bool ok = idx < ce;
        tokLds[lane] = ok ? elist[e * N_TOK + idx] : 0;
        gLds[lane]   = ok ? eg[e * N_TOK + idx]    : 0.f;
    }
    __syncthreads();

    const int ksel = lane >> 4;              // 0 or 1 (which K-half this lane holds)
    const int lnk  = lane & 15;              // M (within half) for A, N for B

    // A fragment sources: rows m=0..15 and m=16..31 of the token tile
    const _Float16* xrow0 = xh + (size_t)tokLds[lnk]      * D_IN + ksel * 16;
    const _Float16* xrow1 = xh + (size_t)tokLds[16 + lnk] * D_IN + ksel * 16;
    // B fragment sources: 4 output subtiles, 16 contiguous K per lane
    const size_t wrow = ((size_t)e * O_OUT + ob + lnk) * D_IN + ksel * 16;
    const _Float16* w0 = wh + wrow;
    const _Float16* w1 = wh + wrow + (size_t)16 * D_IN;
    const _Float16* w2 = wh + wrow + (size_t)32 * D_IN;
    const _Float16* w3 = wh + wrow + (size_t)48 * D_IN;

    v8f c00 = {}, c01 = {}, c02 = {}, c03 = {};   // rows 0..15  x 4 subtiles
    v8f c10 = {}, c11 = {}, c12 = {}, c13 = {};   // rows 16..31 x 4 subtiles
    for (int d0 = 0; d0 < D_IN; d0 += 32) {
        v8h a0lo = *(const v8h*)(xrow0 + d0);
        v8h a0hi = *(const v8h*)(xrow0 + d0 + 8);
        v8h a1lo = *(const v8h*)(xrow1 + d0);
        v8h a1hi = *(const v8h*)(xrow1 + d0 + 8);
        v16h a0, a1;
        #pragma unroll
        for (int j = 0; j < 8; ++j) {
            a0[j] = a0lo[j]; a0[8 + j] = a0hi[j];
            a1[j] = a1lo[j]; a1[8 + j] = a1hi[j];
        }
        v16h b0 = *(const v16h*)(w0 + d0);
        v16h b1 = *(const v16h*)(w1 + d0);
        v16h b2 = *(const v16h*)(w2 + d0);
        v16h b3 = *(const v16h*)(w3 + d0);
        c00 = __builtin_amdgcn_wmma_f32_16x16x32_f16(false, a0, false, b0, (short)0, c00, false, false);
        c10 = __builtin_amdgcn_wmma_f32_16x16x32_f16(false, a1, false, b0, (short)0, c10, false, false);
        c01 = __builtin_amdgcn_wmma_f32_16x16x32_f16(false, a0, false, b1, (short)0, c01, false, false);
        c11 = __builtin_amdgcn_wmma_f32_16x16x32_f16(false, a1, false, b1, (short)0, c11, false, false);
        c02 = __builtin_amdgcn_wmma_f32_16x16x32_f16(false, a0, false, b2, (short)0, c02, false, false);
        c12 = __builtin_amdgcn_wmma_f32_16x16x32_f16(false, a1, false, b2, (short)0, c12, false, false);
        c03 = __builtin_amdgcn_wmma_f32_16x16x32_f16(false, a0, false, b3, (short)0, c03, false, false);
        c13 = __builtin_amdgcn_wmma_f32_16x16x32_f16(false, a1, false, b3, (short)0, c13, false, false);
    }

    // Epilogue: y[tok, o] += gate * (acc + bias). Each (token,expert) pair is
    // unique per list entry; a token appears in exactly 2 expert lists.
    const int rowbase = ksel * 8;            // C/D layout: M = v + 8*(lane>=16)
    const float b0s = eb[(size_t)e * O_OUT + ob + 0  + lnk];
    const float b1s = eb[(size_t)e * O_OUT + ob + 16 + lnk];
    const float b2s = eb[(size_t)e * O_OUT + ob + 32 + lnk];
    const float b3s = eb[(size_t)e * O_OUT + ob + 48 + lnk];
    #pragma unroll
    for (int v = 0; v < 8; ++v) {
        int m0 = rowbase + v;                // rows 0..15 half
        if (mt * 32 + m0 < ce) {
            float g = gLds[m0];
            size_t yb = (size_t)tokLds[m0] * O_OUT + ob + lnk;
            atomicAdd(&y[yb +  0], g * (c00[v] + b0s));
            atomicAdd(&y[yb + 16], g * (c01[v] + b1s));
            atomicAdd(&y[yb + 32], g * (c02[v] + b2s));
            atomicAdd(&y[yb + 48], g * (c03[v] + b3s));
        }
        int m1 = 16 + rowbase + v;           // rows 16..31 half
        if (mt * 32 + m1 < ce) {
            float g = gLds[m1];
            size_t yb = (size_t)tokLds[m1] * O_OUT + ob + lnk;
            atomicAdd(&y[yb +  0], g * (c10[v] + b0s));
            atomicAdd(&y[yb + 16], g * (c11[v] + b1s));
            atomicAdd(&y[yb + 32], g * (c12[v] + b2s));
            atomicAdd(&y[yb + 48], g * (c13[v] + b3s));
        }
    }
}

// ------------------------------------------------------------------
extern "C" void kernel_launch(void* const* d_in, const int* in_sizes, int n_in,
                              void* d_out, int out_size, void* d_ws, size_t ws_size,
                              hipStream_t stream) {
    const float* x     = (const float*)d_in[0];   // [N, D]
    const float* wg    = (const float*)d_in[1];   // [D, E]
    const float* ew    = (const float*)d_in[2];   // [E, O, D]
    const float* ebias = (const float*)d_in[3];   // [E, O]
    const float* sw    = (const float*)d_in[4];   // [O, D]
    float* out = (float*)d_out;                   // y [N,O] then loss scalar

    char* ws = (char*)d_ws;
    int*      cnt   = (int*)(ws + WS_CNT);
    float*    imp   = (float*)(ws + WS_IMP);
    float*    loadv = (float*)(ws + WS_LOAD);
    int*      elist = (int*)(ws + WS_ELIST);
    float*    egw   = (float*)(ws + WS_EG);
    _Float16* xh    = (_Float16*)(ws + WS_XH);
    _Float16* whp   = (_Float16*)(ws + WS_WH);

    // y: 2048*3072 floats = 1572864 float4s
    moe_zero_kernel<<<6144, 256, 0, stream>>>(out, cnt, imp, loadv);
    // Xh: 2048*768/8 = 196608 threads
    moe_cvt_x_kernel<<<768, 256, 0, stream>>>(x, xh);
    // Wh: 8*3072*768/8 = 2359296 threads
    moe_cvt_w_kernel<<<9216, 256, 0, stream>>>(ew, sw, whp);
    // gating: one block per token
    moe_gate_kernel<<<N_TOK, 256, 0, stream>>>(x, wg, cnt, imp, loadv, elist, egw);
    // loss scalar at out[N*O]
    moe_loss_kernel<<<1, 1, 0, stream>>>(imp, loadv, out + (size_t)N_TOK * O_OUT);
    // sparse expert GEMM: (O/64, maxTokenTiles(32), E)
    dim3 grid(O_OUT / 64, N_TOK / 32, E_EXP);
    moe_gemm_kernel<<<grid, 32, 0, stream>>>(xh, whp, ebias, cnt, elist, egw, out);
}
